// GAT_51788715655929
// MI455X (gfx1250) — matrix operand order repaired and checked by reading, not compiled, not run on hardware
//
#include <hip/hip_runtime.h>

#define D 64            // feature dim (in = hid = out = 64)
#define ROWS_PER_BLK 128
#define XSTRIDE 72      // padded LDS row stride (bf16 elems): 144B, conflict-free
#define WSTRIDE 72

typedef __attribute__((ext_vector_type(16))) __bf16 v16bf;
typedef __attribute__((ext_vector_type(8)))  float  v8f;

union FragB16 { v16bf v; uint4 q[2]; };

// ---------------- small helpers ----------------

__device__ __forceinline__ void atomicMaxF(float* addr, float val) {
    // monotonic-bit-pattern trick; works for mixed signs, init -inf
    if (val >= 0.0f) atomicMax((int*)addr, __float_as_int(val));
    else             atomicMin((unsigned int*)addr, __float_as_uint(val));
}

// vdst[i] = sum_o Wdst[i][o] * attdst[o]   (folds dst projection into a vector)
__global__ void fuse_att_kernel(const float* __restrict__ Wdst,
                                const float* __restrict__ attdst,
                                float* __restrict__ vdst) {
    int i = threadIdx.x;  // 64 threads
    float s = 0.0f;
    for (int o = 0; o < D; ++o) s += Wdst[i * D + o] * attdst[o];
    vdst[i] = s;
}

__global__ void init_layer_kernel(float* __restrict__ acc,
                                  float* __restrict__ segmax,
                                  float* __restrict__ segsum, int N) {
    int i = blockIdx.x * blockDim.x + threadIdx.x;
    if (i < N * D) acc[i] = 0.0f;
    if (i < N) { segmax[i] = -__builtin_huge_valf(); segsum[i] = 0.0f; }
}

// ---------------- WMMA projection GEMM ----------------
// Hsrc = bf16(X) @ bf16(Wsrc), Lin = bf16(X) @ bf16(Wlin)  (fp32 accumulate),
// Asrc[r] = Hsrc[r,:] . attsrc,  Adst[r] = X[r,:] . vdst
__global__ __launch_bounds__(256) void gat_gemm_kernel(
    const float* __restrict__ X,       // [N,64]
    const float* __restrict__ Wsrc,    // [64,64] (k-major: W[k][n])
    const float* __restrict__ Wlin,    // [64,64]
    const float* __restrict__ attsrc,  // [64]
    const float* __restrict__ vdst,    // [64]
    float* __restrict__ Hsrc,          // [N,64]
    float* __restrict__ Lin,           // [N,64]
    float* __restrict__ Asrc,          // [N]
    float* __restrict__ Adst,          // [N]
    int N)
{
    __shared__ __align__(32) __bf16 sX [ROWS_PER_BLK * XSTRIDE]; // 18 KB
    __shared__ __align__(32) __bf16 sWsT[D * WSTRIDE];           // 9 KB, transposed
    __shared__ __align__(32) __bf16 sWlT[D * WSTRIDE];           // 9 KB, transposed
    __shared__ float sAtt[D];
    __shared__ float sVd[D];

    const int tid = threadIdx.x;
    const int rowBase = blockIdx.x * ROWS_PER_BLK;

    // stage weights transposed: sWT[n][k] = W[k][n]  (B-fragments contiguous)
    for (int i = tid; i < D * D; i += 256) {
        int k = i >> 6, n = i & 63;
        sWsT[n * WSTRIDE + k] = (__bf16)Wsrc[i];
        sWlT[n * WSTRIDE + k] = (__bf16)Wlin[i];
    }
    if (tid < D) { sAtt[tid] = attsrc[tid]; sVd[tid] = vdst[tid]; }
    // stage 128x64 tile of X (coalesced reads), zero-pad tail rows
    for (int i = tid; i < ROWS_PER_BLK * D; i += 256) {
        int r = i >> 6, c = i & 63;
        float v = (rowBase + r < N) ? X[(rowBase + r) * D + c] : 0.0f;
        sX[r * XSTRIDE + c] = (__bf16)v;
    }
    __syncthreads();

    // a_dst for this tile's 128 rows (whole waves 0..3 -> EXEC uniform per wave)
    if (tid < ROWS_PER_BLK) {
        int r = rowBase + tid;
        if (r < N) {
            float s = 0.0f;
            for (int k = 0; k < D; ++k) s += (float)sX[tid * XSTRIDE + k] * sVd[k];
            Adst[r] = s;
        }
    }

    const int wave  = tid >> 5;       // 8 waves, 16 rows each
    const int lane  = tid & 31;
    const int m0    = wave * 16;
    const int M     = lane & 15;      // lanes 0-15 / 16-31 both map M=0..15
    const int khalf = lane >> 4;

    // A tiles (16x32 bf16, wave32, contiguous k order used by this code): two K-chunks of 32.
    // Per lane: elems 0-15 at k=c*32+khalf*16+0..15 (contiguous) -> two 16B loads.
    FragB16 A[2];
    for (int c = 0; c < 2; ++c) {
        int abase = (m0 + M) * XSTRIDE + c * 32 + khalf * 16;  // 16B aligned
        A[c].q[0] = *(const uint4*)&sX[abase];
        A[c].q[1] = *(const uint4*)&sX[abase + 8];
    }

    const v8f vzero = {0.f,0.f,0.f,0.f,0.f,0.f,0.f,0.f};
    v8f Cs[4], Cl[4];
    for (int t = 0; t < 4; ++t) { Cs[t] = vzero; Cl[t] = vzero; }

    for (int t = 0; t < 4; ++t) {           // 4 column tiles of 16
        const int n = t * 16 + M;
        for (int c = 0; c < 2; ++c) {       // K chunks
            // B 32x16 layout: lane's 16 elems are k = c*32 + khalf*16 + 0..15,
            // contiguous in the transposed LDS image -> two 16B loads.
            int bbase = n * WSTRIDE + c * 32 + khalf * 16;     // 16B aligned
            FragB16 Bs, Bl;
            Bs.q[0] = *(const uint4*)&sWsT[bbase];
            Bs.q[1] = *(const uint4*)&sWsT[bbase + 8];
            Bl.q[0] = *(const uint4*)&sWlT[bbase];
            Bl.q[1] = *(const uint4*)&sWlT[bbase + 8];
            Cs[t] = __builtin_amdgcn_wmma_f32_16x16x32_bf16(
                        false, A[c].v, false, Bs.v, (short)0, Cs[t], false, false);
            Cl[t] = __builtin_amdgcn_wmma_f32_16x16x32_bf16(
                        false, A[c].v, false, Bl.v, (short)0, Cl[t], false, false);
        }
    }

    // write-out + fused a_src = Hsrc . attsrc
    float p[8];
    for (int r = 0; r < 8; ++r) p[r] = 0.0f;
    for (int t = 0; t < 4; ++t) {
        const int n = t * 16 + M;
        const float att = sAtt[n];
        for (int r = 0; r < 8; ++r) {
            int row = rowBase + m0 + khalf * 8 + r;   // C layout: M = r + 8*(lane/16)
            float hs = Cs[t][r];
            if (row < N) {
                Hsrc[row * D + n] = hs;
                Lin[row * D + n]  = Cl[t][r];
            }
            p[r] += hs * att;
        }
    }
    // reduce across the 16 lanes that share a row-half
    for (int r = 0; r < 8; ++r) {
        float v = p[r];
        v += __shfl_xor(v, 1, 32);
        v += __shfl_xor(v, 2, 32);
        v += __shfl_xor(v, 4, 32);
        v += __shfl_xor(v, 8, 32);
        if (M == 0) {
            int row = rowBase + m0 + khalf * 8 + r;
            if (row < N) Asrc[row] = v;
        }
    }
}

// ---------------- edge passes ----------------

__global__ void edge_score_kernel(const int* __restrict__ src,
                                  const int* __restrict__ dst,
                                  const float* __restrict__ asrc,
                                  const float* __restrict__ adst,
                                  float* __restrict__ escore,
                                  float* __restrict__ segmax, int E) {
    int i = blockIdx.x * blockDim.x + threadIdx.x;
    if (i >= E) return;
    int s = src[i], d = dst[i];
    float e = asrc[s] + adst[d];
    e = (e >= 0.0f) ? e : 0.2f * e;       // leaky relu, slope 0.2
    escore[i] = e;
    atomicMaxF(&segmax[d], e);
}

__global__ void edge_exp_kernel(const int* __restrict__ dst,
                                const float* __restrict__ segmax,
                                float* __restrict__ escore,   // in: e, out: w
                                float* __restrict__ segsum, int E) {
    int i = blockIdx.x * blockDim.x + threadIdx.x;
    if (i >= E) return;
    int d = dst[i];
    float w = __expf(escore[i] - segmax[d]);
    escore[i] = w;
    atomicAdd(&segsum[d], w);
}

// one wave per edge, 2 features per lane: coalesced gather + cacheline atomics
__global__ void edge_scatter_kernel(const int* __restrict__ src,
                                    const int* __restrict__ dst,
                                    const float* __restrict__ w,
                                    const float* __restrict__ segsum,
                                    const float* __restrict__ hsrc,
                                    float* __restrict__ acc, int E) {
    long long g = (long long)blockIdx.x * blockDim.x + threadIdx.x;
    int edge = (int)(g >> 5);
    int lane = (int)(g & 31);
    if (edge >= E) return;
    int s = src[edge], d = dst[edge];
    float alpha = w[edge] / segsum[d];
    atomicAdd(&acc[d * D + lane],      hsrc[s * D + lane]      * alpha);
    atomicAdd(&acc[d * D + lane + 32], hsrc[s * D + lane + 32] * alpha);
}

__global__ void epilogue_kernel(const float* __restrict__ acc,
                                const float* __restrict__ lin,
                                const float* __restrict__ b,
                                const float* __restrict__ bl,
                                float* __restrict__ out, int total) {
    int i = blockIdx.x * blockDim.x + threadIdx.x;
    if (i >= total) return;
    int c = i & 63;
    float v = acc[i] + b[c] + lin[i] + bl[c];
    out[i] = v > 0.0f ? v : 0.0f;
}

// ---------------- orchestration ----------------

extern "C" void kernel_launch(void* const* d_in, const int* in_sizes, int n_in,
                              void* d_out, int out_size, void* d_ws, size_t ws_size,
                              hipStream_t stream) {
    const float* x    = (const float*)d_in[0];
    const int*   ei   = (const int*)d_in[1];
    const float* W1s  = (const float*)d_in[2];
    const float* W1d  = (const float*)d_in[3];
    const float* at1s = (const float*)d_in[4];
    const float* at1d = (const float*)d_in[5];
    const float* b1   = (const float*)d_in[6];
    const float* Wl1  = (const float*)d_in[7];
    const float* bl1  = (const float*)d_in[8];
    const float* W2s  = (const float*)d_in[9];
    const float* W2d  = (const float*)d_in[10];
    const float* at2s = (const float*)d_in[11];
    const float* at2d = (const float*)d_in[12];
    const float* b2   = (const float*)d_in[13];
    const float* Wl2  = (const float*)d_in[14];
    const float* bl2  = (const float*)d_in[15];

    const int N = in_sizes[0] / D;
    const int E = in_sizes[1] / 2;
    const int* src = ei;
    const int* dst = ei + E;

    // workspace layout (floats)
    float* ws   = (float*)d_ws;
    size_t nd   = (size_t)N * D;
    float* hsrc = ws;
    float* lin  = hsrc + nd;
    float* acc  = lin + nd;
    float* h    = acc + nd;
    float* esc  = h + nd;          // E floats
    float* asrc = esc + E;
    float* adst = asrc + N;
    float* smax = adst + N;
    float* ssum = smax + N;
    float* vdst = ssum + N;        // 64 floats

    const int gemmBlocks = (N + ROWS_PER_BLK - 1) / ROWS_PER_BLK;
    const int ndBlocks   = (int)((nd + 255) / 256);
    const int eBlocks    = (E + 255) / 256;
    const int scBlocks   = (int)(((long long)E * 32 + 255) / 256);

    // ---- layer 1 ----
    fuse_att_kernel<<<1, 64, 0, stream>>>(W1d, at1d, vdst);
    init_layer_kernel<<<ndBlocks, 256, 0, stream>>>(acc, smax, ssum, N);
    gat_gemm_kernel<<<gemmBlocks, 256, 0, stream>>>(x, W1s, Wl1, at1s, vdst,
                                                    hsrc, lin, asrc, adst, N);
    edge_score_kernel<<<eBlocks, 256, 0, stream>>>(src, dst, asrc, adst, esc, smax, E);
    edge_exp_kernel<<<eBlocks, 256, 0, stream>>>(dst, smax, esc, ssum, E);
    edge_scatter_kernel<<<scBlocks, 256, 0, stream>>>(src, dst, esc, ssum, hsrc, acc, E);
    epilogue_kernel<<<ndBlocks, 256, 0, stream>>>(acc, lin, b1, bl1, h, (int)nd);

    // ---- layer 2 ----
    fuse_att_kernel<<<1, 64, 0, stream>>>(W2d, at2d, vdst);
    init_layer_kernel<<<ndBlocks, 256, 0, stream>>>(acc, smax, ssum, N);
    gat_gemm_kernel<<<gemmBlocks, 256, 0, stream>>>(h, W2s, Wl2, at2s, vdst,
                                                    hsrc, lin, asrc, adst, N);
    edge_score_kernel<<<eBlocks, 256, 0, stream>>>(src, dst, asrc, adst, esc, smax, E);
    edge_exp_kernel<<<eBlocks, 256, 0, stream>>>(dst, smax, esc, ssum, E);
    edge_scatter_kernel<<<scBlocks, 256, 0, stream>>>(src, dst, esc, ssum, hsrc, acc, E);
    epilogue_kernel<<<ndBlocks, 256, 0, stream>>>(acc, lin, b2, bl2, (float*)d_out, (int)nd);
}
